// PolicyLSTM_64742337020096
// MI455X (gfx1250) — compile-verified, matrix operand order checked by reading, not run on hardware
//
#include <hip/hip_runtime.h>
#include <math.h>

// ---------------------------------------------------------------------------
// Types for CDNA5 WMMA (wave32)
// ---------------------------------------------------------------------------
typedef __attribute__((ext_vector_type(16))) __bf16 v16bf;
typedef __attribute__((ext_vector_type(8)))  __bf16 bf16x8;
typedef __attribute__((ext_vector_type(8)))  float  v8f;

__device__ inline v16bf cat16(bf16x8 lo, bf16x8 hi) {
    return __builtin_shufflevector(lo, hi, 0,1,2,3,4,5,6,7,8,9,10,11,12,13,14,15);
}

__device__ inline void cluster_barrier() {
    asm volatile("" ::: "memory");
    __builtin_amdgcn_s_cluster_barrier();   // s_barrier_signal -3 ; s_barrier_wait -3
    asm volatile("" ::: "memory");
}

// slice column s in [0,256): gate = s/64, jj = s%64 -> weight row index
__device__ inline int row_of(int s, int wj) {
    return (s >> 6) * 1024 + wj * 64 + (s & 63);
}

// ---------------------------------------------------------------------------
// Tensor Data Mover: 1-D bf16 tile (nelem 2-byte elements) global -> LDS.
// D# group0: count=1 | lds_addr | global_addr(57b) | type=2
// D# group1: data_size=2B, tile_dim0=nelem, tensor_dim0=nelem (1-D tile)
// Issued by ONE wave (TDM ignores EXEC; TENSORcnt is per-wave).
// This toolchain exposes the 6-arg builtin:
//   (u32x4 g0, i32x8 g1, i32x4 g2, i32x4 g3, i32x8 extra, i32 cpol)
// ---------------------------------------------------------------------------
#if defined(__has_builtin)
#if __has_builtin(__builtin_amdgcn_tensor_load_to_lds)
#define HAVE_TDM 1
#endif
#endif

#ifdef HAVE_TDM
__device__ inline void tdm_load_1d_bf16(void* lds_dst, const void* gsrc,
                                        unsigned nelem /* <= 65535 */) {
    typedef unsigned int u32x4 __attribute__((ext_vector_type(4)));
    typedef int          i32x8 __attribute__((ext_vector_type(8)));
    typedef int          i32x4 __attribute__((ext_vector_type(4)));
    unsigned           lds = (unsigned)(unsigned long long)lds_dst;
    unsigned long long ga  = (unsigned long long)gsrc;
    u32x4 g0;
    g0[0] = 1u;                                            // count=1 (user D#)
    g0[1] = lds;                                           // lds_addr (bytes)
    g0[2] = (unsigned)ga;                                  // global_addr[31:0]
    g0[3] = ((unsigned)(ga >> 32) & 0x01FFFFFFu)           // global_addr[56:32]
            | 0x80000000u;                                 // type = 2 ("image")
    i32x8 g1;
    g1[0] = 0x00010000;                                    // data_size=1 (2B)
    g1[1] = (int)((nelem & 0xFFFFu) << 16);                // tensor_dim0[15:0]
    g1[2] = (int)(nelem >> 16);                            // tensor_dim0[31:16]
    g1[3] = (int)((nelem & 0xFFFFu) << 16);                // tile_dim0 (16b)
    g1[4] = 0;                                             // tile_dim1/2 unused
    g1[5] = (int)nelem;                                    // dim0 stride (n/a 1-D)
    g1[6] = 0;
    g1[7] = 0;
    i32x4 z4 = {0, 0, 0, 0};
    i32x8 z8 = {0, 0, 0, 0, 0, 0, 0, 0};
    __builtin_amdgcn_tensor_load_to_lds(g0, g1, z4, z4, z8, 0); // TENSORcnt++
}
#endif

// ---------------------------------------------------------------------------
// fp32 -> bf16 weight conversion (one-time; weights then live in L2)
// ---------------------------------------------------------------------------
__global__ void f32_to_bf16_kernel(const float* __restrict__ src,
                                   __bf16* __restrict__ dst, int n) {
    for (int i = blockIdx.x * blockDim.x + threadIdx.x; i < n;
         i += gridDim.x * blockDim.x)
        dst[i] = (__bf16)src[i];
}

// ---------------------------------------------------------------------------
// RMSNorm-gated LSTM scan.
// Grid = 64 WGs = 4 clusters(batch chunks of 16 rows) x 16 WGs(64-wide j-slice).
// Each WG per step: z[16,256] = x_t @ Wx^T + h @ Wh^T + b  via bf16 WMMA,
// then gate-RMSNorm (cluster-wide reduction), c update, c-RMSNorm
// (cluster-wide reduction), h update. 3 cluster barriers per step.
// h chunk staged to LDS per step via Tensor Data Mover.
// ---------------------------------------------------------------------------
template <int KX, bool IS_L0>
__global__ __launch_bounds__(256, 1)
void scan_kernel(const float*  __restrict__ x_f32,   // L0: obs [64,512,256] fp32
                 const __bf16* __restrict__ x_bf,    // L1: hs0 [64,512,1024] bf16
                 const __bf16* __restrict__ Wxb,     // [4096, KX] bf16
                 const __bf16* __restrict__ Whb,     // [4096, 1024] bf16
                 const float*  __restrict__ bias,    // [4096]
                 const float*  __restrict__ gN,      // [4,1024]
                 const float*  __restrict__ cN,      // [1024]
                 __bf16*       __restrict__ hcur,    // [64,1024] bf16 (state)
                 __bf16*       __restrict__ hs_out,  // L0: [64,512,1024] bf16
                 float*        __restrict__ red)     // [4][512][16][16][5]
{
    constexpr int T = 512, H = 1024, M = 16, J = 64, CW = 16, NS = 256;
    const int tid  = threadIdx.x;
    const int lane = tid & 31;
    const int wv   = tid >> 5;               // 8 waves
    const int cb   = blockIdx.x >> 4;        // batch-chunk / cluster index
    const int wj   = blockIdx.x & 15;        // j-slice within cluster
    const int m0   = cb * M;

    __shared__ __bf16 h_lds[M][H];           // 32 KB  (staged h chunk, TDM dest)
    __shared__ __bf16 x_lds[M][256];         //  8 KB  (L0 staged x_t, bf16)
    __shared__ float  z_lds[M][NS];          // 16 KB  (gate pre-activations)
    __shared__ float  c_lds[M][J];           //  4 KB  (persistent cell state)
    __shared__ float  rinv_s[M][4];
    __shared__ float  crinv_s[M];

    // ---- init: c = 0, own slice of h = 0 ----
    for (int i = tid; i < M * J; i += 256) c_lds[i >> 6][i & 63] = 0.f;
    for (int i = tid; i < M * J; i += 256)
        hcur[(size_t)(m0 + (i >> 6)) * H + wj * J + (i & 63)] = (__bf16)0.f;
    __syncthreads();
    __threadfence();
    cluster_barrier();

    const int r_a  = lane & 15;              // A-matrix row / B-matrix column
    const int off  = (lane < 16) ? 0 : 8;    // documented 16-bit K swizzle
    const int madd = (lane < 16) ? 0 : 8;    // C-matrix M offset

    for (int t = 0; t < T; ++t) {
        // ---- stage h chunk [16,1024] bf16 into LDS ----
#ifdef HAVE_TDM
        if (wv == 0) {                       // one DMA op per WG (per-wave cnt)
            tdm_load_1d_bf16(&h_lds[0][0], hcur + (size_t)m0 * H,
                             (unsigned)(M * H));
            __builtin_amdgcn_s_wait_tensorcnt(0);
        }
#else
        {
            const uint4* src = (const uint4*)(hcur + (size_t)m0 * H);
            uint4* dst = (uint4*)(&h_lds[0][0]);
            for (int i = tid; i < (M * H) / 8; i += 256) dst[i] = src[i];
        }
#endif
        if (IS_L0) {  // stage+convert obs tile fp32 -> bf16
            for (int i = tid; i < M * 256; i += 256) {
                int b = i >> 8, k = i & 255;
                x_lds[b][k] = (__bf16)x_f32[((size_t)(m0 + b) * T + t) * 256 + k];
            }
        }
        __syncthreads();                     // publishes TDM-staged h to all waves

        const __bf16* xrow =
            IS_L0 ? (const __bf16*)nullptr
                  : (x_bf + ((size_t)(m0 + r_a) * T + t) * H);

        // ---- GEMM: z[16,256] += x@Wx^T + h@Wh^T + bias, bf16 WMMA ----
        for (int nt2 = 0; nt2 < 2; ++nt2) {
            const int ntile = wv + nt2 * 8;              // 16 n-tiles / 8 waves
            const int s     = ntile * 16 + (lane & 15);  // slice column
            const int wrow  = row_of(s, wj);             // weight row
            const __bf16* bx = Wxb + (size_t)wrow * KX;
            const __bf16* bh = Whb + (size_t)wrow * H;
            v8f acc = {0.f, 0.f, 0.f, 0.f, 0.f, 0.f, 0.f, 0.f};

            for (int kb = 0; kb < KX; kb += 32) {        // x contribution
                bf16x8 alo, ahi;
                if (IS_L0) {
                    alo = *(const bf16x8*)&x_lds[r_a][kb + off];
                    ahi = *(const bf16x8*)&x_lds[r_a][kb + off + 16];
                } else {
                    alo = *(const bf16x8*)(xrow + kb + off);
                    ahi = *(const bf16x8*)(xrow + kb + off + 16);
                }
                bf16x8 blo = *(const bf16x8*)(bx + kb + off);
                bf16x8 bhi = *(const bf16x8*)(bx + kb + off + 16);
                acc = __builtin_amdgcn_wmma_f32_16x16x32_bf16(
                    false, cat16(alo, ahi), false, cat16(blo, bhi),
                    (short)0, acc, false, false);
            }
            for (int kb = 0; kb < H; kb += 32) {         // h contribution
                __builtin_prefetch(bh + kb + 256, 0, 1); // global_prefetch_b8
                bf16x8 alo = *(const bf16x8*)&h_lds[r_a][kb + off];
                bf16x8 ahi = *(const bf16x8*)&h_lds[r_a][kb + off + 16];
                bf16x8 blo = *(const bf16x8*)(bh + kb + off);
                bf16x8 bhi = *(const bf16x8*)(bh + kb + off + 16);
                acc = __builtin_amdgcn_wmma_f32_16x16x32_bf16(
                    false, cat16(alo, ahi), false, cat16(blo, bhi),
                    (short)0, acc, false, false);
            }
            const float bv = bias[wrow];
            for (int r = 0; r < 8; ++r) z_lds[r + madd][s] = acc[r] + bv;
        }
        __syncthreads();

        // ---- per-(b,gate) sum of squares partial -> global slot ----
        const size_t rbase = (((size_t)cb * T + t) * CW + wj) * M * 5;
        if (tid < 64) {
            int b = tid >> 2, g = tid & 3;
            float ss = 0.f;
            for (int jj = 0; jj < J; ++jj) {
                float v = z_lds[b][g * J + jj];
                ss += v * v;
            }
            red[rbase + (size_t)b * 5 + g] = ss;
        }
        __syncthreads();
        __threadfence();
        cluster_barrier();                                  // barrier A

        if (tid < 64) {  // deterministic reduction of 16 partials
            int b = tid >> 2, g = tid & 3;
            float tot = 0.f;
            volatile const float* vr = red;
            for (int w2 = 0; w2 < CW; ++w2)
                tot += vr[((((size_t)cb * T + t) * CW + w2) * M + b) * 5 + g];
            rinv_s[b][g] = rsqrtf(tot * (1.f / 1024.f) + 1e-6f);
        }
        __syncthreads();

        // ---- gates, cell update; stash c^2 for cnorm ----
        float c4[4], o4[4];
        for (int e = 0; e < 4; ++e) {
            int idx = tid + e * 256;
            int b = idx >> 6, jj = idx & 63;
            int j = wj * J + jj;
            float zi = z_lds[b][jj]         * rinv_s[b][0] * gN[0 * H + j];
            float zf = z_lds[b][J + jj]     * rinv_s[b][1] * gN[1 * H + j];
            float zg = z_lds[b][2 * J + jj] * rinv_s[b][2] * gN[2 * H + j];
            float zo = z_lds[b][3 * J + jj] * rinv_s[b][3] * gN[3 * H + j];
            float iv = 1.f / (1.f + expf(-zi));
            float fv = 1.f / (1.f + expf(-zf));
            float gv = tanhf(zg);
            float ov = 1.f / (1.f + expf(-zo));
            float c  = fv * c_lds[b][jj] + iv * gv;
            c_lds[b][jj] = c;
            c4[e] = c; o4[e] = ov;
            z_lds[b][jj] = c * c;   // only this thread reads/writes this slot
        }
        __syncthreads();
        if (tid < 16) {
            float ss = 0.f;
            for (int jj = 0; jj < J; ++jj) ss += z_lds[tid][jj];
            red[rbase + (size_t)tid * 5 + 4] = ss;
        }
        __syncthreads();
        __threadfence();
        cluster_barrier();                                  // barrier B
        if (tid < 16) {
            float tot = 0.f;
            volatile const float* vr = red;
            for (int w2 = 0; w2 < CW; ++w2)
                tot += vr[((((size_t)cb * T + t) * CW + w2) * M + tid) * 5 + 4];
            crinv_s[tid] = rsqrtf(tot * (1.f / 1024.f) + 1e-6f);
        }
        __syncthreads();

        // ---- h = o * tanh(rmsnorm(c)); publish slice ----
        for (int e = 0; e < 4; ++e) {
            int idx = tid + e * 256;
            int b = idx >> 6, jj = idx & 63;
            int j = wj * J + jj;
            float hval = o4[e] * tanhf(c4[e] * crinv_s[b] * cN[j]);
            __bf16 hb = (__bf16)hval;
            hcur[(size_t)(m0 + b) * H + j] = hb;
            if (IS_L0) hs_out[((size_t)(m0 + b) * T + t) * H + j] = hb;
        }
        __syncthreads();
        __threadfence();
        cluster_barrier();                                  // barrier C
    }
}

// ---------------------------------------------------------------------------
// Head: mean = h_last @ Wout^T + bout ; std = softplus(std_param)+0.01
// ---------------------------------------------------------------------------
__global__ void head_kernel(const __bf16* __restrict__ h1,
                            const float* __restrict__ Wout,
                            const float* __restrict__ bout,
                            const float* __restrict__ stdp,
                            float* __restrict__ out) {
    int tid = blockIdx.x * blockDim.x + threadIdx.x;
    if (tid < 512) {
        int b = tid >> 3, a = tid & 7;
        float acc = bout[a];
        const __bf16* hr = h1 + (size_t)b * 1024;
        const float*  wr = Wout + (size_t)a * 1024;
        for (int k = 0; k < 1024; ++k) acc += (float)hr[k] * wr[k];
        out[tid] = acc;
    } else if (tid < 520) {
        int i = tid - 512;
        float x = stdp[i];
        float sp = (x > 20.f) ? x : log1pf(expf(x));
        out[512 + i] = sp + 0.01f;
    }
}

// ---------------------------------------------------------------------------
extern "C" void kernel_launch(void* const* d_in, const int* in_sizes, int n_in,
                              void* d_out, int out_size, void* d_ws, size_t ws_size,
                              hipStream_t stream) {
    (void)in_sizes; (void)n_in; (void)out_size; (void)ws_size;
    const float* obs  = (const float*)d_in[0];
    const float* Wih0 = (const float*)d_in[1];
    const float* Whh0 = (const float*)d_in[2];
    const float* b0   = (const float*)d_in[3];
    const float* gn0  = (const float*)d_in[4];
    const float* cn0  = (const float*)d_in[5];
    const float* Wih1 = (const float*)d_in[6];
    const float* Whh1 = (const float*)d_in[7];
    const float* b1   = (const float*)d_in[8];
    const float* gn1  = (const float*)d_in[9];
    const float* cn1  = (const float*)d_in[10];
    const float* Wout = (const float*)d_in[11];
    const float* bout = (const float*)d_in[12];
    const float* stdp = (const float*)d_in[13];
    float* out = (float*)d_out;

    // workspace layout (all sizes 256B-aligned)
    char* w = (char*)d_ws;
    size_t o = 0;
    __bf16* Wih0b = (__bf16*)(w + o); o += (size_t)4096 * 256  * 2;  //  2 MB
    __bf16* Whh0b = (__bf16*)(w + o); o += (size_t)4096 * 1024 * 2;  //  8 MB
    __bf16* Wih1b = (__bf16*)(w + o); o += (size_t)4096 * 1024 * 2;  //  8 MB
    __bf16* Whh1b = (__bf16*)(w + o); o += (size_t)4096 * 1024 * 2;  //  8 MB
    __bf16* hcur  = (__bf16*)(w + o); o += (size_t)64 * 1024 * 2;    // 128 KB
    __bf16* hs0   = (__bf16*)(w + o); o += (size_t)64 * 512 * 1024 * 2; // 64 MB
    float*  red   = (float*) (w + o); o += (size_t)4 * 512 * 16 * 16 * 5 * 4; // 10 MB

    // one-time weight conversion fp32 -> bf16 (then resident in 192MB L2)
    f32_to_bf16_kernel<<<2048, 256, 0, stream>>>(Wih0, Wih0b, 4096 * 256);
    f32_to_bf16_kernel<<<4096, 256, 0, stream>>>(Whh0, Whh0b, 4096 * 1024);
    f32_to_bf16_kernel<<<4096, 256, 0, stream>>>(Wih1, Wih1b, 4096 * 1024);
    f32_to_bf16_kernel<<<4096, 256, 0, stream>>>(Whh1, Whh1b, 4096 * 1024);

    // layer 0: x = obs (fp32, converted in-kernel), writes full h sequence
    scan_kernel<256, true><<<64, 256, 0, stream>>>(
        obs, nullptr, Wih0b, Whh0b, b0, gn0, cn0, hcur, hs0, red);
    // layer 1: x = hs0 (bf16), only final h needed
    scan_kernel<1024, false><<<64, 256, 0, stream>>>(
        nullptr, hs0, Wih1b, Whh1b, b1, gn1, cn1, hcur, nullptr, red);

    head_kernel<<<3, 256, 0, stream>>>(hcur, Wout, bout, stdp, out);
}
